// MambaBlock_53137335386571
// MI455X (gfx1250) — compile-verified, matrix operand order checked by reading, not run on hardware
//
#include <hip/hip_runtime.h>
#include <cstdint>

// ---------------------------------------------------------------------------
// MambaBlock for MI455X (gfx1250): wave32, WMMA f16->f32, TDM-staged GEMMs.
// B=8 H=W=64 C=256 DIN=256 DS=16 DTR=16 K=4 ; NTOK = 32768
// ---------------------------------------------------------------------------

#define NTOK   32768
#define CDIM   256
#define DIN    256
#define DS     16
#define DTR    16
#define SEQ    64
#define NSEQ   512

typedef __attribute__((ext_vector_type(16))) _Float16 v16h;
typedef __attribute__((ext_vector_type(8)))  float    v8f;

#if defined(__gfx1250__) && __has_builtin(__builtin_amdgcn_tensor_load_to_lds)
#define HAVE_TDM 1
#else
#define HAVE_TDM 0
#endif

#if HAVE_TDM
typedef unsigned int v4u __attribute__((ext_vector_type(4)));
typedef int v4i __attribute__((ext_vector_type(4)));
typedef int v8i __attribute__((ext_vector_type(8)));

#if __has_builtin(__builtin_amdgcn_s_wait_tensorcnt)
#define WAIT_TENSORCNT(n) __builtin_amdgcn_s_wait_tensorcnt(n)
#else
#define WAIT_TENSORCNT(n) asm volatile("s_wait_tensorcnt %0" ::"i"(n))
#endif

// Issue one TDM 2-D tile load: rows x (elems_per_row halves) from global
// (row stride = row_stride_elems halves) into contiguous LDS at lds_byte_addr.
// D# layout per CDNA5 ISA ch.8: group0 = {flags, lds_addr, global_addr, type=2},
// group1 = {data_size=2B, tensor dims, tile dims, strides}.
__device__ __forceinline__ void tdm_load_2d(uint32_t lds_byte_addr,
                                            const void* gptr,
                                            uint32_t elems_per_row,
                                            uint32_t rows,
                                            uint64_t row_stride_elems) {
    uint64_t ga = (uint64_t)(uintptr_t)gptr;
    uint32_t tw = (uint32_t)row_stride_elems;   // tensor_dim0 (>= tile_dim0)
    uint32_t th = rows;                         // tensor_dim1
    v4u g0;
    g0[0] = 1u;                                        // count=1, user mode
    g0[1] = lds_byte_addr;                             // lds_addr
    g0[2] = (uint32_t)ga;                              // global_addr[31:0]
    g0[3] = ((uint32_t)(ga >> 32) & 0x01FFFFFFu) | 0x80000000u; // addr[56:32]|type=2
    v8i g1;
    g1[0] = (int)(1u << 16);                           // data_size = 2 bytes
    g1[1] = (int)((tw & 0xFFFFu) << 16);               // tensor_dim0[15:0]
    g1[2] = (int)(((tw >> 16) & 0xFFFFu) | ((th & 0xFFFFu) << 16));
    g1[3] = (int)(((th >> 16) & 0xFFFFu) | (elems_per_row << 16)); // tile_dim0
    g1[4] = (int)(rows & 0xFFFFu);                     // tile_dim1, tile_dim2=0
    g1[5] = (int)(uint32_t)row_stride_elems;           // dim0_stride[31:0]
    g1[6] = (int)((uint32_t)(row_stride_elems >> 32) & 0xFFFFu);
    g1[7] = 0;
    v4i g2 = {1, 1, 0, 0};                             // tensor_dim2/3 = 1
    v4i g3 = {0, 0, 0, 0};
#if defined(__clang_major__) && __clang_major__ >= 23
    v8i gz = {0, 0, 0, 0, 0, 0, 0, 0};
    __builtin_amdgcn_tensor_load_to_lds(g0, g1, g2, g3, gz, 0);
#else
    __builtin_amdgcn_tensor_load_to_lds(g0, g1, g2, g3, 0);
#endif
}
#endif  // HAVE_TDM

// 16x32 f16 WMMA fragment load: lane row already applied, element i ->
// k = (i>=8?16:0) + (i&7); khalf supplies the (lane>=16?8:0) shift.
__device__ __forceinline__ v16h frag16(const _Float16* p) {
    union { v16h v; uint4 q[2]; } u;
    u.q[0] = *(const uint4*)(p);
    u.q[1] = *(const uint4*)(p + 16);
    return u.v;
}

__device__ __forceinline__ v8f wmma_f16(v16h a, v16h b, v8f c) {
    return __builtin_amdgcn_wmma_f32_16x16x32_f16(false, a, false, b, (short)0, c,
                                                  false, false);
}

__device__ __forceinline__ float silu_f(float x) { return x / (1.0f + __expf(-x)); }
__device__ __forceinline__ float softplus_f(float x) {
    return (x > 20.0f) ? x : log1pf(__expf(x));
}

// One 32x64 wave-tile MMA step over a 32-wide K slice held in LDS.
__device__ __forceinline__ void mma_block(const _Float16* curA, const _Float16* curB,
                                          v8f (&acc)[2][4], int wm, int wn,
                                          int mrow, int khalf) {
    v16h a[2], b[4];
#pragma unroll
    for (int mi = 0; mi < 2; ++mi)
        a[mi] = frag16(curA + (wm * 32 + mi * 16 + mrow) * 32 + khalf);
#pragma unroll
    for (int ni = 0; ni < 4; ++ni)
        b[ni] = frag16(curB + (wn * 64 + ni * 16 + mrow) * 32 + khalf);
#pragma unroll
    for (int mi = 0; mi < 2; ++mi)
#pragma unroll
        for (int ni = 0; ni < 4; ++ni)
            acc[mi][ni] = wmma_f16(a[mi], b[ni], acc[mi][ni]);
}

// ---------------------------------------------------------------------------
__global__ void cvt_kernel(const float* __restrict__ src, _Float16* __restrict__ dst,
                           int n) {
    int i = blockIdx.x * 256 + threadIdx.x;
    if (i < n) dst[i] = (_Float16)src[i];
}

// LayerNorm: one wave per token, 8 channels per lane.
__global__ void ln_kernel(const float* __restrict__ x, const float* __restrict__ g,
                          const float* __restrict__ b, _Float16* __restrict__ xn) {
    int wv = threadIdx.x >> 5, lane = threadIdx.x & 31;
    size_t tok = (size_t)blockIdx.x * 8 + wv;
    const float4* xr = (const float4*)(x + tok * CDIM);
    float4 v0 = xr[lane];
    float4 v1 = xr[lane + 32];
    float s = v0.x + v0.y + v0.z + v0.w + v1.x + v1.y + v1.z + v1.w;
    float q = v0.x*v0.x + v0.y*v0.y + v0.z*v0.z + v0.w*v0.w
            + v1.x*v1.x + v1.y*v1.y + v1.z*v1.z + v1.w*v1.w;
#pragma unroll
    for (int off = 16; off > 0; off >>= 1) {
        s += __shfl_xor(s, off);
        q += __shfl_xor(q, off);
    }
    float mu  = s * (1.0f / CDIM);
    float var = q * (1.0f / CDIM) - mu * mu;
    float rs  = rsqrtf(var + 1e-6f);
    const float4* gp = (const float4*)g;
    const float4* bp = (const float4*)b;
    float4 g0 = gp[lane], g1 = gp[lane + 32];
    float4 b0 = bp[lane], b1 = bp[lane + 32];
    _Float16* o = xn + tok * CDIM;
    int c0 = 4 * lane, c1 = 4 * lane + 128;
    o[c0 + 0] = (_Float16)((v0.x - mu) * rs * g0.x + b0.x);
    o[c0 + 1] = (_Float16)((v0.y - mu) * rs * g0.y + b0.y);
    o[c0 + 2] = (_Float16)((v0.z - mu) * rs * g0.z + b0.z);
    o[c0 + 3] = (_Float16)((v0.w - mu) * rs * g0.w + b0.w);
    o[c1 + 0] = (_Float16)((v1.x - mu) * rs * g1.x + b1.x);
    o[c1 + 1] = (_Float16)((v1.y - mu) * rs * g1.y + b1.y);
    o[c1 + 2] = (_Float16)((v1.z - mu) * rs * g1.z + b1.z);
    o[c1 + 3] = (_Float16)((v1.w - mu) * rs * g1.w + b1.w);
}

// ---------------------------------------------------------------------------
// WMMA GEMM: OUT(MxN) = A(MxK f16 rowmajor) * Bw(NxK f16 rowmajor)^T
// 128x128 block tile, 8 waves (4x2), double-buffered TDM tile staging.
// EPI: 0 = f16 store ; 1 = +bias, exact GELU, f16 ; 2 = +bias +resid, f32.
// ---------------------------------------------------------------------------
template <int EPI>
__global__ void __launch_bounds__(256)
gemm_kernel(const _Float16* __restrict__ A, const _Float16* __restrict__ Bw,
            int M, int N, int K, const float* __restrict__ bias,
            const float* __restrict__ resid, void* __restrict__ out) {
    __shared__ __align__(16) char smem_raw[32768];
    _Float16* sA = (_Float16*)smem_raw;              // [2][128*32]
    _Float16* sB = (_Float16*)(smem_raw + 16384);    // [2][128*32]
    _Float16* sOut = (_Float16*)smem_raw;            // 128x128 after k-loop

    int tid = threadIdx.x, wv = tid >> 5, lane = tid & 31;
    int m0 = blockIdx.x * 128, n0 = blockIdx.y * 128;
    int wm = wv & 3, wn = wv >> 2;
    int khalf = (lane >= 16) ? 8 : 0;
    int mrow = lane & 15;
    v8f acc[2][4] = {};
    int nk = K >> 5;

#if HAVE_TDM
    uint32_t ldsA = (uint32_t)(uintptr_t)sA;
    uint32_t ldsB = (uint32_t)(uintptr_t)sB;
    if (wv == 0) {
        tdm_load_2d(ldsA, A + (size_t)m0 * K, 32, 128, (uint64_t)K);
        tdm_load_2d(ldsB, Bw + (size_t)n0 * K, 32, 128, (uint64_t)K);
    }
    for (int ki = 0; ki < nk; ++ki) {
        int buf = ki & 1;
        if (wv == 0) {
            if (ki + 1 < nk) {
                tdm_load_2d(ldsA + (buf ^ 1) * 8192,
                            A + (size_t)m0 * K + (ki + 1) * 32, 32, 128, (uint64_t)K);
                tdm_load_2d(ldsB + (buf ^ 1) * 8192,
                            Bw + (size_t)n0 * K + (ki + 1) * 32, 32, 128, (uint64_t)K);
                WAIT_TENSORCNT(2);   // the two loads of the current buffer
            } else {
                WAIT_TENSORCNT(0);
            }
        }
        __syncthreads();
        mma_block(sA + buf * 4096, sB + buf * 4096, acc, wm, wn, mrow, khalf);
        __syncthreads();
    }
#else
    for (int k0 = 0; k0 < K; k0 += 32) {
#pragma unroll
        for (int idx = tid; idx < 512; idx += 256) {
            int row = idx >> 2, kv = (idx & 3) << 3;
            *(uint4*)(sA + row * 32 + kv) =
                *(const uint4*)(A + (size_t)(m0 + row) * K + k0 + kv);
            *(uint4*)(sB + row * 32 + kv) =
                *(const uint4*)(Bw + (size_t)(n0 + row) * K + k0 + kv);
        }
        __syncthreads();
        mma_block(sA, sB, acc, wm, wn, mrow, khalf);
        __syncthreads();
    }
#endif

    int rbase = (lane >= 16) ? 8 : 0;
    if (EPI == 2) {
        // f32 output with bias + residual; half-wave writes 64B segments.
#pragma unroll
        for (int mi = 0; mi < 2; ++mi)
#pragma unroll
            for (int ni = 0; ni < 4; ++ni)
#pragma unroll
                for (int r = 0; r < 8; ++r) {
                    int row = m0 + wm * 32 + mi * 16 + rbase + r;
                    int col = n0 + wn * 64 + ni * 16 + (lane & 15);
                    size_t oidx = (size_t)row * N + col;
                    ((float*)out)[oidx] = acc[mi][ni][r] + bias[col] + resid[oidx];
                }
    } else {
        // Stage f16 tile in LDS, then coalesced 16B global stores.
#pragma unroll
        for (int mi = 0; mi < 2; ++mi)
#pragma unroll
            for (int ni = 0; ni < 4; ++ni)
#pragma unroll
                for (int r = 0; r < 8; ++r) {
                    int rloc = wm * 32 + mi * 16 + rbase + r;
                    int cloc = wn * 64 + ni * 16 + (lane & 15);
                    float v = acc[mi][ni][r];
                    if (EPI == 1) {
                        v += bias[n0 + cloc];
                        v = 0.5f * v * (1.0f + erff(v * 0.70710678118654752f));
                    }
                    sOut[rloc * 128 + cloc] = (_Float16)v;
                }
        __syncthreads();
        _Float16* o16 = (_Float16*)out;
#pragma unroll
        for (int idx = tid; idx < 2048; idx += 256) {
            int row = idx >> 4, vv = (idx & 15) << 3;
            *(uint4*)(o16 + (size_t)(m0 + row) * N + n0 + vv) =
                *(const uint4*)(sOut + row * 128 + vv);
        }
    }
}

// ---------------------------------------------------------------------------
// Fused per-sequence Mamba core: one block (8 waves) per length-64 sequence.
//  TDM u-tile load -> conv+SiLU -> xproj WMMA (64x48) -> selective scan
//  -> out-proj WMMA (64x256) staged in LDS -> coalesced store into concat buf.
// Dynamic LDS: u 32KB + us 32KB + dbc 12KB = 76KB (CDNA5 WGP has 320KB).
// ---------------------------------------------------------------------------
__global__ void __launch_bounds__(256)
scan_kernel(const _Float16* __restrict__ uz,       // NTOK x 512
            const _Float16* __restrict__ xproj16,  // 48 x 256
            const float* __restrict__ conv_w, const float* __restrict__ conv_b,
            const float* __restrict__ dt_w, const float* __restrict__ dt_b,
            const float* __restrict__ A_log, const float* __restrict__ Dp,
            const _Float16* __restrict__ outw16,   // 256 x 256
            _Float16* __restrict__ fused,          // NTOK x 512
            int col_off, int vertical) {
    extern __shared__ char smem[];
    _Float16* lds_u  = (_Float16*)smem;              // 64x256 raw u, later ys
    _Float16* lds_us = (_Float16*)(smem + 32768);    // 64x256 silu(conv(u))
    float*    lds_dbc = (float*)(smem + 65536);      // 64x48 {dt,B,C}

    int tid = threadIdx.x, wv = tid >> 5, lane = tid & 31;
    int sidx = blockIdx.x;
    int b = sidx >> 6, r = sidx & 63;
    int tok_base = vertical ? (b * 4096 + r) : (b * 4096 + r * 64);
    int stride   = vertical ? 64 : 1;

    // 1. u tile (64 steps x 256 ch, row stride 512 or 32768 halves) via TDM.
#if HAVE_TDM
    if (wv == 0) {
        tdm_load_2d((uint32_t)(uintptr_t)lds_u, uz + (size_t)tok_base * 512,
                    DIN, SEQ, (uint64_t)stride * 512);
        WAIT_TENSORCNT(0);
    }
#else
#pragma unroll 4
    for (int t = 0; t < SEQ; ++t)
        lds_u[t * DIN + tid] = uz[(size_t)(tok_base + t * stride) * 512 + tid];
#endif
    __syncthreads();

    // 2. depthwise causal conv (K=4) + SiLU; thread = channel d.
    {
        int d = tid;
        float w0 = conv_w[d * 4 + 0], w1 = conv_w[d * 4 + 1];
        float w2 = conv_w[d * 4 + 2], w3 = conv_w[d * 4 + 3];
        float cb = conv_b[d];
        float um3 = 0.f, um2 = 0.f, um1 = 0.f;
#pragma unroll 4
        for (int t = 0; t < SEQ; ++t) {
            float u0 = (float)lds_u[t * DIN + d];
            float a = cb + w0 * um3 + w1 * um2 + w2 * um1 + w3 * u0;
            lds_us[t * DIN + d] = (_Float16)silu_f(a);
            um3 = um2; um2 = um1; um1 = u0;
        }
    }
    __syncthreads();

    // 3. dbc = us @ xproj^T : 4 x 3 WMMA tiles over 8 waves.
    {
        int khalf = (lane >= 16) ? 8 : 0;
        int mrow = lane & 15;
        for (int tile = wv; tile < 12; tile += 8) {
            int mt = tile & 3, nt = tile >> 2;
            v8f acc = {};
#pragma unroll
            for (int k0 = 0; k0 < DIN; k0 += 32) {
                v16h a  = frag16(lds_us + (mt * 16 + mrow) * DIN + k0 + khalf);
                v16h bb = frag16(xproj16 + (nt * 16 + mrow) * DIN + k0 + khalf);
                acc = wmma_f16(a, bb, acc);
            }
            int rbase = (lane >= 16) ? 8 : 0;
#pragma unroll
            for (int rr = 0; rr < 8; ++rr)
                lds_dbc[(mt * 16 + rbase + rr) * 48 + nt * 16 + (lane & 15)] = acc[rr];
        }
    }
    __syncthreads();

    // 4. selective scan: thread = channel d, 16 recurrent states in registers.
    {
        int d = tid;
        float Ar[DS], dw[DTR];
#pragma unroll
        for (int s = 0; s < DS; ++s) Ar[s] = -__expf(A_log[d * DS + s]);
#pragma unroll
        for (int s = 0; s < DTR; ++s) dw[s] = dt_w[d * DTR + s];
        float dtb = dt_b[d], Dd = Dp[d];
        float h[DS];
#pragma unroll
        for (int s = 0; s < DS; ++s) h[s] = 0.f;

        for (int t = 0; t < SEQ; ++t) {
            const float* dbc_t = lds_dbc + t * 48;
            float dtv = dtb;
#pragma unroll
            for (int s = 0; s < DTR; ++s) dtv += dbc_t[s] * dw[s];
            float delta = softplus_f(dtv);
            float u = (float)lds_us[t * DIN + d];
            float du = delta * u;
            float y = 0.f;
#pragma unroll
            for (int s = 0; s < DS; ++s) {
                float dA = __expf(delta * Ar[s]);
                h[s] = dA * h[s] + du * dbc_t[16 + s];
                y += h[s] * dbc_t[32 + s];
            }
            y += u * Dd;
            float z = (float)uz[(size_t)(tok_base + t * stride) * 512 + 256 + d];
            y *= silu_f(z);
            lds_u[t * DIN + d] = (_Float16)y;   // reuse raw-u LDS as ys
        }
    }
    __syncthreads();

    // 5. out-proj: ys(64x256) @ out_w^T; stage result tile in lds_us.
    {
        int khalf = (lane >= 16) ? 8 : 0;
        int mrow = lane & 15;
#pragma unroll
        for (int j = 0; j < 8; ++j) {
            int tile = wv * 8 + j;
            int mt = tile & 3, nt = tile >> 2;
            v8f acc = {};
#pragma unroll
            for (int k0 = 0; k0 < DIN; k0 += 32) {
                v16h a  = frag16(lds_u + (mt * 16 + mrow) * DIN + k0 + khalf);
                v16h bb = frag16(outw16 + (nt * 16 + mrow) * DIN + k0 + khalf);
                acc = wmma_f16(a, bb, acc);
            }
            int rbase = (lane >= 16) ? 8 : 0;
#pragma unroll
            for (int rr = 0; rr < 8; ++rr)
                lds_us[(mt * 16 + rbase + rr) * DIN + nt * 16 + (lane & 15)] =
                    (_Float16)acc[rr];
        }
    }
    __syncthreads();

    // 6. coalesced 16B stores into the concat (fused) buffer.
#pragma unroll
    for (int idx = tid; idx < SEQ * 32; idx += 256) {
        int t = idx >> 5, vv = (idx & 31) << 3;
        *(uint4*)(fused + (size_t)(tok_base + t * stride) * 512 + col_off + vv) =
            *(const uint4*)(lds_us + t * DIN + vv);
    }
}

// ---------------------------------------------------------------------------
// Host launcher
// ---------------------------------------------------------------------------
extern "C" void kernel_launch(void* const* d_in, const int* in_sizes, int n_in,
                              void* d_out, int out_size, void* d_ws, size_t ws_size,
                              hipStream_t stream) {
    (void)in_sizes; (void)n_in; (void)out_size; (void)ws_size;

    const float* x        = (const float*)d_in[0];
    const float* gamma    = (const float*)d_in[1];
    const float* beta     = (const float*)d_in[2];
    const float* mh_in_w  = (const float*)d_in[3];
    const float* mh_convw = (const float*)d_in[4];
    const float* mh_convb = (const float*)d_in[5];
    const float* mh_xproj = (const float*)d_in[6];
    const float* mh_dtw   = (const float*)d_in[7];
    const float* mh_dtb   = (const float*)d_in[8];
    const float* mh_Alog  = (const float*)d_in[9];
    const float* mh_D     = (const float*)d_in[10];
    const float* mh_outw  = (const float*)d_in[11];
    const float* mv_in_w  = (const float*)d_in[12];
    const float* mv_convw = (const float*)d_in[13];
    const float* mv_convb = (const float*)d_in[14];
    const float* mv_xproj = (const float*)d_in[15];
    const float* mv_dtw   = (const float*)d_in[16];
    const float* mv_dtb   = (const float*)d_in[17];
    const float* mv_Alog  = (const float*)d_in[18];
    const float* mv_D     = (const float*)d_in[19];
    const float* mv_outw  = (const float*)d_in[20];
    const float* fw1      = (const float*)d_in[21];
    const float* fb1      = (const float*)d_in[22];
    const float* fw2      = (const float*)d_in[23];
    const float* fb2      = (const float*)d_in[24];

    char* w = (char*)d_ws;
    size_t off = 0;
    auto alloc = [&](size_t bytes) -> void* {
        void* p = w + off;
        off = (off + bytes + 255) & ~(size_t)255;
        return p;
    };
    _Float16* xn      = (_Float16*)alloc((size_t)NTOK * CDIM * 2);
    _Float16* uz_h    = (_Float16*)alloc((size_t)NTOK * 512 * 2);
    _Float16* uz_v    = (_Float16*)alloc((size_t)NTOK * 512 * 2);
    _Float16* fusedb  = (_Float16*)alloc((size_t)NTOK * 512 * 2);
    _Float16* hidden  = (_Float16*)alloc((size_t)NTOK * CDIM * 2);
    _Float16* w_in_h  = (_Float16*)alloc(512 * 256 * 2);
    _Float16* w_in_v  = (_Float16*)alloc(512 * 256 * 2);
    _Float16* xprj_h  = (_Float16*)alloc(48 * 256 * 2);
    _Float16* xprj_v  = (_Float16*)alloc(48 * 256 * 2);
    _Float16* outw_h  = (_Float16*)alloc(256 * 256 * 2);
    _Float16* outw_v  = (_Float16*)alloc(256 * 256 * 2);
    _Float16* fw1_16  = (_Float16*)alloc(256 * 512 * 2);
    _Float16* fw2_16  = (_Float16*)alloc(256 * 256 * 2);

    auto cvt = [&](const float* s, _Float16* d, int n) {
        cvt_kernel<<<(n + 255) / 256, 256, 0, stream>>>(s, d, n);
    };
    cvt(mh_in_w, w_in_h, 512 * 256);
    cvt(mv_in_w, w_in_v, 512 * 256);
    cvt(mh_xproj, xprj_h, 48 * 256);
    cvt(mv_xproj, xprj_v, 48 * 256);
    cvt(mh_outw, outw_h, 256 * 256);
    cvt(mv_outw, outw_v, 256 * 256);
    cvt(fw1, fw1_16, 256 * 512);
    cvt(fw2, fw2_16, 256 * 256);

    ln_kernel<<<NTOK / 8, 256, 0, stream>>>(x, gamma, beta, xn);

    gemm_kernel<0><<<dim3(NTOK / 128, 4), 256, 0, stream>>>(
        xn, w_in_h, NTOK, 512, 256, nullptr, nullptr, uz_h);
    gemm_kernel<0><<<dim3(NTOK / 128, 4), 256, 0, stream>>>(
        xn, w_in_v, NTOK, 512, 256, nullptr, nullptr, uz_v);

    const int SCAN_LDS = 32768 + 32768 + 64 * 48 * 4;
    hipFuncSetAttribute((const void*)scan_kernel,
                        hipFuncAttributeMaxDynamicSharedMemorySize, SCAN_LDS);
    scan_kernel<<<NSEQ, 256, SCAN_LDS, stream>>>(
        uz_h, xprj_h, mh_convw, mh_convb, mh_dtw, mh_dtb, mh_Alog, mh_D,
        outw_h, fusedb, /*col_off=*/0, /*vertical=*/0);
    scan_kernel<<<NSEQ, 256, SCAN_LDS, stream>>>(
        uz_v, xprj_v, mv_convw, mv_convb, mv_dtw, mv_dtb, mv_Alog, mv_D,
        outw_v, fusedb, /*col_off=*/256, /*vertical=*/1);

    gemm_kernel<1><<<dim3(NTOK / 128, 2), 256, 0, stream>>>(
        fusedb, fw1_16, NTOK, 256, 512, fb1, nullptr, hidden);
    gemm_kernel<2><<<dim3(NTOK / 128, 2), 256, 0, stream>>>(
        hidden, fw2_16, NTOK, 256, 256, fb2, x, d_out);
}
